// MultiHeadAttention_28338194219533
// MI455X (gfx1250) — matrix operand order repaired and checked by reading, not compiled, not run on hardware
//
#include <hip/hip_runtime.h>
#include <hip/hip_bf16.h>

// MHA forward for MI455X (gfx1250, wave32, WMMA).
// Pipeline: f32->bf16 converts/transposes -> QKV GEMM (WMMA bf16) ->
// register-resident flash attention (WMMA bf16, online softmax) ->
// output GEMM + bias (WMMA bf16, f32 accum/store).
//
// Dims: B=8, tokens N=1024/batch (M=8192 total), C=768, heads=12, hd=64, 3C=2304.

typedef __attribute__((ext_vector_type(16))) __bf16 v16bf;
typedef __attribute__((ext_vector_type(8)))  float  v8f;

union FragU {
    v16bf v;
    uint4 q[2];
    unsigned int   w[8];
    unsigned short s[16];
};

__device__ __forceinline__ unsigned short f2bf(float f) {
    unsigned int u = __float_as_uint(f);
    u += 0x7FFFu + ((u >> 16) & 1u);   // round-to-nearest-even
    return (unsigned short)(u >> 16);
}

// A-matrix fragment (16Mx32K bf16). Order used here: lane row = L%16,
// elems 0..15 -> K = k0 + 16*half + e (16 contiguous, matching B).
__device__ __forceinline__ v16bf ld_fragA(const unsigned short* rowptr, int k0, int half) {
    FragU u;
    u.q[0] = *(const uint4*)(rowptr + k0 + half * 16);
    u.q[1] = *(const uint4*)(rowptr + k0 + half * 16 + 8);
    return u.v;
}

// B-matrix fragment (32Kx16N bf16). Order used here: lane col = L%16,
// elems 0..15 -> K = k0 + 16*half + e (16 contiguous). Source stored transposed
// (rowptr indexed by output column n), so loads are contiguous.
__device__ __forceinline__ v16bf ld_fragB(const unsigned short* rowptr, int k0, int half) {
    FragU u;
    u.q[0] = *(const uint4*)(rowptr + k0 + half * 16);
    u.q[1] = *(const uint4*)(rowptr + k0 + half * 16 + 8);
    return u.v;
}

__device__ __forceinline__ v8f wmma_bf16(v16bf a, v16bf b, v8f c) {
    return __builtin_amdgcn_wmma_f32_16x16x32_bf16(false, a, false, b, (short)0, c, false, false);
}

// ---------------------------------------------------------------- prep kernels
__global__ void cvt_bf16_kernel(const float* __restrict__ in,
                                unsigned short* __restrict__ out, int n) {
    int i = blockIdx.x * 256 + threadIdx.x;
    if (i < n) out[i] = f2bf(in[i]);
}

// W [K][N] f32 -> Wt [N][K] bf16
__global__ void transpose_bf16_kernel(const float* __restrict__ in,
                                      unsigned short* __restrict__ out, int K, int N) {
    int idx = blockIdx.x * 256 + threadIdx.x;
    if (idx < K * N) {
        int n = idx % N, k = idx / N;
        out[(size_t)n * K + k] = f2bf(in[idx]);
    }
}

// --------------------------------------------------------------- QKV GEMM
// x_bf [8192][768] @ Wqkv (stored as Wt [2304][768]) -> q [b,h,n,64]*0.125,
// k [b,h,n,64], vT [b,h,64,1024]. 32x64 tile per wave, 8 waves/block.
__global__ __launch_bounds__(256) void qkv_gemm_kernel(
        const unsigned short* __restrict__ xbf,
        const unsigned short* __restrict__ wt,
        unsigned short* __restrict__ qb,
        unsigned short* __restrict__ kb,
        unsigned short* __restrict__ vt) {
    const int lane = threadIdx.x & 31;
    const int col  = lane & 15;
    const int half = lane >> 4;
    const int wid  = blockIdx.x * 8 + (threadIdx.x >> 5);   // 9216 tiles
    const int m0 = (wid / 36) * 32;
    const int n0 = (wid % 36) * 64;

    const unsigned short* arow0 = xbf + (size_t)(m0 + col) * 768;
    const unsigned short* arow1 = xbf + (size_t)(m0 + 16 + col) * 768;
    const unsigned short* brow0 = wt + (size_t)(n0 +  0 + col) * 768;
    const unsigned short* brow1 = wt + (size_t)(n0 + 16 + col) * 768;
    const unsigned short* brow2 = wt + (size_t)(n0 + 32 + col) * 768;
    const unsigned short* brow3 = wt + (size_t)(n0 + 48 + col) * 768;

    v8f acc[2][4] = {};
    for (int k = 0; k < 768; k += 32) {
        v16bf a0 = ld_fragA(arow0, k, half);
        v16bf a1 = ld_fragA(arow1, k, half);
        v16bf b0 = ld_fragB(brow0, k, half);
        v16bf b1 = ld_fragB(brow1, k, half);
        v16bf b2 = ld_fragB(brow2, k, half);
        v16bf b3 = ld_fragB(brow3, k, half);
        acc[0][0] = wmma_bf16(a0, b0, acc[0][0]);
        acc[1][0] = wmma_bf16(a1, b0, acc[1][0]);
        acc[0][1] = wmma_bf16(a0, b1, acc[0][1]);
        acc[1][1] = wmma_bf16(a1, b1, acc[1][1]);
        acc[0][2] = wmma_bf16(a0, b2, acc[0][2]);
        acc[1][2] = wmma_bf16(a1, b2, acc[1][2]);
        acc[0][3] = wmma_bf16(a0, b3, acc[0][3]);
        acc[1][3] = wmma_bf16(a1, b3, acc[1][3]);
    }

    #pragma unroll
    for (int j = 0; j < 4; ++j) {
        const int c = n0 + j * 16 + col;      // output column 0..2303
        const int which = c / 768;            // 0=q 1=k 2=v (uniform per tile)
        const int cc = c % 768;
        const int h = cc >> 6, d = cc & 63;
        #pragma unroll
        for (int i = 0; i < 2; ++i) {
            const int mbase = m0 + i * 16 + half * 8;
            #pragma unroll
            for (int vi = 0; vi < 8; ++vi) {
                const int row = mbase + vi;                 // global token
                const int b = row >> 10, n = row & 1023;
                const size_t hb = (size_t)b * 12 + h;
                const float val = acc[i][j][vi];
                if (which == 0)
                    qb[(hb * 1024 + n) * 64 + d] = f2bf(val * 0.125f); // * hd^-0.5
                else if (which == 1)
                    kb[(hb * 1024 + n) * 64 + d] = f2bf(val);
                else
                    vt[(hb * 64 + d) * 1024 + n] = f2bf(val);
            }
        }
    }
}

// --------------------------------------------------------------- attention
// One 16-row q-tile per wave, flash-attention with online softmax.
// S^T = K * Q^T  (D layout: lane col = q, VGPR rows = k-token)  -> softmax
// per-lane + shfl_xor(16) -> P^T packed into B fragment (xor-16 swap to match
// B's 16-contiguous-K layout) -> O^T += V^T * P^T.
__global__ __launch_bounds__(256) void attn_kernel(
        const unsigned short* __restrict__ qb,
        const unsigned short* __restrict__ kb,
        const unsigned short* __restrict__ vt,
        unsigned short* __restrict__ obf) {
    const int lane = threadIdx.x & 31;
    const int col  = lane & 15;
    const int half = lane >> 4;
    const int wid  = blockIdx.x * 8 + (threadIdx.x >> 5);   // 6144 units
    const int head = wid >> 6;                              // 0..95
    const int q0   = (wid & 63) * 16;

    const unsigned short* Q = qb + (size_t)head * 1024 * 64;
    const unsigned short* K = kb + (size_t)head * 1024 * 64;
    const unsigned short* V = vt + (size_t)head * 64 * 1024;

    const unsigned short* qrow = Q + (size_t)(q0 + col) * 64;
    const v16bf bq0 = ld_fragB(qrow, 0,  half);
    const v16bf bq1 = ld_fragB(qrow, 32, half);

    const unsigned short* vrow0 = V + (size_t)( 0 + col) * 1024;
    const unsigned short* vrow1 = V + (size_t)(16 + col) * 1024;
    const unsigned short* vrow2 = V + (size_t)(32 + col) * 1024;
    const unsigned short* vrow3 = V + (size_t)(48 + col) * 1024;

    v8f acc0 = {}, acc1 = {}, acc2 = {}, acc3 = {};
    float m = -1e30f, l = 0.0f;

    for (int kt = 0; kt < 1024; kt += 32) {
        const unsigned short* krow0 = K + (size_t)(kt + col) * 64;
        const unsigned short* krow1 = K + (size_t)(kt + 16 + col) * 64;
        v8f s0 = {}, s1 = {};
        s0 = wmma_bf16(ld_fragA(krow0, 0,  half), bq0, s0);
        s0 = wmma_bf16(ld_fragA(krow0, 32, half), bq1, s0);
        s1 = wmma_bf16(ld_fragA(krow1, 0,  half), bq0, s1);
        s1 = wmma_bf16(ld_fragA(krow1, 32, half), bq1, s1);

        float mloc = -1e30f;
        #pragma unroll
        for (int i = 0; i < 8; ++i) {
            mloc = fmaxf(mloc, s0[i]);
            mloc = fmaxf(mloc, s1[i]);
        }
        mloc = fmaxf(mloc, __shfl_xor(mloc, 16, 32));
        const float mnew  = fmaxf(m, mloc);
        const float alpha = __expf(m - mnew);

        unsigned int p0w[4], p1w[4];
        float lsum = 0.0f;
        #pragma unroll
        for (int i = 0; i < 4; ++i) {
            float a0 = __expf(s0[2 * i]     - mnew);
            float a1 = __expf(s0[2 * i + 1] - mnew);
            float b0 = __expf(s1[2 * i]     - mnew);
            float b1 = __expf(s1[2 * i + 1] - mnew);
            lsum += a0 + a1 + b0 + b1;
            p0w[i] = (unsigned int)f2bf(a0) | ((unsigned int)f2bf(a1) << 16);
            p1w[i] = (unsigned int)f2bf(b0) | ((unsigned int)f2bf(b1) << 16);
        }
        lsum += __shfl_xor(lsum, 16, 32);
        l = l * alpha + lsum;
        m = mnew;
        acc0 *= alpha; acc1 *= alpha; acc2 *= alpha; acc3 *= alpha;

        // Re-pack S^T D-regs into B fragment of P^T. D lane holds rows
        // 8*half..8*half+7 of each 16-tile; B lane needs K = 16*half + 0..15,
        // so swap halves across lane^16.
        FragU bp;
        #pragma unroll
        for (int i = 0; i < 4; ++i) {
            unsigned int q0w = __shfl_xor(p0w[i], 16, 32);
            unsigned int q1w = __shfl_xor(p1w[i], 16, 32);
            bp.w[i]     = half ? q1w    : p0w[i];
            bp.w[i + 4] = half ? p1w[i] : q0w;
        }

        acc0 = wmma_bf16(ld_fragA(vrow0, kt, half), bp.v, acc0);
        acc1 = wmma_bf16(ld_fragA(vrow1, kt, half), bp.v, acc1);
        acc2 = wmma_bf16(ld_fragA(vrow2, kt, half), bp.v, acc2);
        acc3 = wmma_bf16(ld_fragA(vrow3, kt, half), bp.v, acc3);
    }

    const float invl = 1.0f / l;
    const int b = head / 12, h = head % 12;
    const int token = b * 1024 + q0 + col;                  // lane col = q
    unsigned short* orow = obf + (size_t)token * 768 + h * 64 + half * 8;
    v8f ar[4] = {acc0, acc1, acc2, acc3};
    #pragma unroll
    for (int dt = 0; dt < 4; ++dt) {
        FragU u;
        #pragma unroll
        for (int i = 0; i < 4; ++i) {
            u.w[i] = (unsigned int)f2bf(ar[dt][2 * i] * invl) |
                     ((unsigned int)f2bf(ar[dt][2 * i + 1] * invl) << 16);
        }
        *(uint4*)(orow + dt * 16) = u.q[0];
    }
}

// --------------------------------------------------------------- out proj
// o_bf [8192][768] @ Wout (stored Wt [768][768]) + b_out -> f32 out [8192][768]
__global__ __launch_bounds__(256) void out_gemm_kernel(
        const unsigned short* __restrict__ obf,
        const unsigned short* __restrict__ wt,
        const float* __restrict__ bias,
        float* __restrict__ out) {
    const int lane = threadIdx.x & 31;
    const int col  = lane & 15;
    const int half = lane >> 4;
    const int wid  = blockIdx.x * 8 + (threadIdx.x >> 5);   // 3072 tiles
    const int m0 = (wid / 12) * 32;
    const int n0 = (wid % 12) * 64;

    const unsigned short* arow0 = obf + (size_t)(m0 + col) * 768;
    const unsigned short* arow1 = obf + (size_t)(m0 + 16 + col) * 768;
    const unsigned short* brow0 = wt + (size_t)(n0 +  0 + col) * 768;
    const unsigned short* brow1 = wt + (size_t)(n0 + 16 + col) * 768;
    const unsigned short* brow2 = wt + (size_t)(n0 + 32 + col) * 768;
    const unsigned short* brow3 = wt + (size_t)(n0 + 48 + col) * 768;

    v8f acc[2][4] = {};
    for (int k = 0; k < 768; k += 32) {
        v16bf a0 = ld_fragA(arow0, k, half);
        v16bf a1 = ld_fragA(arow1, k, half);
        v16bf b0 = ld_fragB(brow0, k, half);
        v16bf b1 = ld_fragB(brow1, k, half);
        v16bf b2 = ld_fragB(brow2, k, half);
        v16bf b3 = ld_fragB(brow3, k, half);
        acc[0][0] = wmma_bf16(a0, b0, acc[0][0]);
        acc[1][0] = wmma_bf16(a1, b0, acc[1][0]);
        acc[0][1] = wmma_bf16(a0, b1, acc[0][1]);
        acc[1][1] = wmma_bf16(a1, b1, acc[1][1]);
        acc[0][2] = wmma_bf16(a0, b2, acc[0][2]);
        acc[1][2] = wmma_bf16(a1, b2, acc[1][2]);
        acc[0][3] = wmma_bf16(a0, b3, acc[0][3]);
        acc[1][3] = wmma_bf16(a1, b3, acc[1][3]);
    }

    #pragma unroll
    for (int j = 0; j < 4; ++j) {
        const int c = n0 + j * 16 + col;
        const float bz = bias[c];
        #pragma unroll
        for (int i = 0; i < 2; ++i) {
            const int mbase = m0 + i * 16 + half * 8;
            #pragma unroll
            for (int vi = 0; vi < 8; ++vi)
                out[(size_t)(mbase + vi) * 768 + c] = acc[i][j][vi] + bz;
        }
    }
}

// ---------------------------------------------------------------- launcher
extern "C" void kernel_launch(void* const* d_in, const int* in_sizes, int n_in,
                              void* d_out, int out_size, void* d_ws, size_t ws_size,
                              hipStream_t stream) {
    (void)in_sizes; (void)n_in; (void)out_size; (void)ws_size;
    const float* x    = (const float*)d_in[0];   // [8,32,32,768]
    const float* Wqkv = (const float*)d_in[1];   // [768,2304]
    const float* Wout = (const float*)d_in[2];   // [768,768]
    const float* bout = (const float*)d_in[3];   // [768]
    float* out = (float*)d_out;

    char* ws = (char*)d_ws;                       // ~68 MB of scratch used
    unsigned short* xbf   = (unsigned short*)(ws + 0);          // 8192*768 bf16
    unsigned short* wqkvt = (unsigned short*)(ws + 12582912);   // 2304*768 bf16
    unsigned short* woutt = (unsigned short*)(ws + 16121856);   // 768*768  bf16
    unsigned short* qb    = (unsigned short*)(ws + 17301504);   // 96*1024*64 bf16
    unsigned short* kb    = (unsigned short*)(ws + 29884416);   // 96*1024*64 bf16
    unsigned short* vt    = (unsigned short*)(ws + 42467328);   // 96*64*1024 bf16
    unsigned short* obf   = (unsigned short*)(ws + 55050240);   // 8192*768 bf16

    cvt_bf16_kernel<<<(8192 * 768 + 255) / 256, 256, 0, stream>>>(x, xbf, 8192 * 768);
    transpose_bf16_kernel<<<(768 * 2304 + 255) / 256, 256, 0, stream>>>(Wqkv, wqkvt, 768, 2304);
    transpose_bf16_kernel<<<(768 * 768 + 255) / 256, 256, 0, stream>>>(Wout, woutt, 768, 768);

    qkv_gemm_kernel<<<1152, 256, 0, stream>>>(xbf, wqkvt, qb, kb, vt);   // 9216 tiles
    attn_kernel<<<768, 256, 0, stream>>>(qb, kb, vt, obf);               // 6144 q-tiles
    out_gemm_kernel<<<384, 256, 0, stream>>>(obf, woutt, bout, out);     // 3072 tiles
}
